// WindowAttention_42331197669435
// MI455X (gfx1250) — matrix operand order repaired and checked by reading, not compiled, not run on hardware
//
#include <hip/hip_runtime.h>
#include <hip/hip_bf16.h>
#include <stdint.h>

// ---------------------------------------------------------------------------
// Swin window attention for MI455X (gfx1250), bf16 WMMA path.
// ---------------------------------------------------------------------------

typedef __attribute__((ext_vector_type(16))) __bf16 v16bf;
typedef __attribute__((ext_vector_type(8)))  float  v8f;

#define N_TOK     49
#define DIM_      256
#define NHEAD     8
#define HDIM      32
#define QKV_COLS  768
#define BTOT      2048
#define NWIN      64
#define NN        (N_TOK * N_TOK)      // 2401
#define ATT_SCALE 0.1767766952966369f  // 1/sqrt(32)

// gfx1250 async global->LDS copies (ASYNCcnt-tracked), guarded so absent
// builtins (or the host pass) fall back to plain vector loads/stores.
// Probe-learned signature: (v4i AS1*, v4i AS3*, imm offset, imm cpol).
#if defined(__gfx1250__) &&                                         \
    __has_builtin(__builtin_amdgcn_global_load_async_to_lds_b128) && \
    __has_builtin(__builtin_amdgcn_s_wait_asynccnt)
#define ASYNC_COPY 1
typedef int v4i_ __attribute__((vector_size(4 * sizeof(int))));
typedef __attribute__((address_space(1))) v4i_* as1_v4ip;
typedef __attribute__((address_space(3))) v4i_* as3_v4ip;
#define ASYNC_B128(g, l) \
  __builtin_amdgcn_global_load_async_to_lds_b128((as1_v4ip)(g), (as3_v4ip)(l), 0, 0)
#define ASYNC_WAIT() __builtin_amdgcn_s_wait_asynccnt(0)
#else
#define ASYNC_COPY 0
#define ASYNC_B128(g, l) (*(uint4*)(l) = *(const uint4*)(g))
#define ASYNC_WAIT()
#endif

union FragU { v16bf v; uint32_t u[8]; };

// A-matrix 16x32 bf16 fragment (ISA 7.12.2): lane L holds row M=L%16;
// lanes 0-15 hold K={0..7,16..23}, lanes 16-31 hold K={8..15,24..31}.
__device__ __forceinline__ v16bf load_frag_a(const __bf16* base, int strideH, int lane) {
  const int m   = lane & 15;
  const int sel = (lane >> 4) & 1;
  const uint32_t* p = (const uint32_t*)(base + m * strideH);
  FragU f;
  const int o0 = sel * 4;       // dwords (sel*8 halfs)
  const int o1 = 8 + sel * 4;   // dwords (16 + sel*8 halfs)
  f.u[0] = p[o0 + 0]; f.u[1] = p[o0 + 1]; f.u[2] = p[o0 + 2]; f.u[3] = p[o0 + 3];
  f.u[4] = p[o1 + 0]; f.u[5] = p[o1 + 1]; f.u[6] = p[o1 + 2]; f.u[7] = p[o1 + 3];
  return f.v;
}

// B-matrix 32x16 bf16 fragment: lane L holds column N=L%16;
// lanes 0-15 hold K={0..7,16..23}, lanes 16-31 hold K={8..15,24..31}.
// baseT points at a [N][K] (column-major-by-N) LDS tile.
__device__ __forceinline__ v16bf load_frag_b(const __bf16* baseT, int strideH, int lane) {
  const int n  = lane & 15;
  const int sel = (lane >> 4) & 1;
  const uint32_t* p = (const uint32_t*)(baseT + n * strideH);
  FragU f;
#pragma unroll
  for (int i = 0; i < 4; ++i) { f.u[i] = p[sel * 4 + i]; f.u[4 + i] = p[8 + sel * 4 + i]; }
  return f.v;
}

// ---------------------------------------------------------------------------
// fp32 -> bf16 conversion (weights)
// ---------------------------------------------------------------------------
__global__ void cvt_f32_bf16(const float* __restrict__ s, __bf16* __restrict__ d, int n) {
  int i = blockIdx.x * 256 + threadIdx.x;
  if (i < n) d[i] = (__bf16)s[i];
}

// ---------------------------------------------------------------------------
// Combined additive bias: cb[w][h][ij] = bias_table[rel_index[ij]][h] + mask[w][ij]
// (64 * 8 * 2401 floats ~= 4.9 MB, L2-resident during attention)
// ---------------------------------------------------------------------------
__global__ void build_bias(const float* __restrict__ bias_table,
                           const int* __restrict__ rel_index,
                           const float* __restrict__ mask,
                           float* __restrict__ cb) {
  int idx = blockIdx.x * 256 + threadIdx.x;
  if (idx < NWIN * NHEAD * NN) {
    const int ij = idx % NN;
    const int wh = idx / NN;
    const int h  = wh & (NHEAD - 1);
    const int w  = wh / NHEAD;
    cb[idx] = bias_table[rel_index[ij] * NHEAD + h] + mask[w * NN + ij];
  }
}

// ---------------------------------------------------------------------------
// 64x64-tile GEMM:  Out[M,Nout] = A[M,K] * Bw[K,Nout] + bias[Nout]
// A is fp32 (converted on the fly) or bf16; Bw is bf16; Out is bf16 or fp32.
// 256 threads = 8 waves; each wave owns a 16x32 output patch (2 WMMA accs).
// M must be a multiple of 64 (100352 = 1568*64 here).
// ---------------------------------------------------------------------------
template <bool A_BF16, bool OUT_BF16>
__global__ __launch_bounds__(256)
void gemm64(const void* __restrict__ Aptr, const __bf16* __restrict__ Bw,
            const float* __restrict__ bias, void* __restrict__ Outp,
            int K, int Nout) {
  __shared__ alignas(16) __bf16 aT[64 * 40];   // [m][k], stride 40 halfs
  __shared__ alignas(16) __bf16 bT[64 * 40];   // [n][k] (transposed), stride 40 halfs

  const int t    = threadIdx.x;
  const int m0   = blockIdx.y * 64;
  const int n0   = blockIdx.x * 64;
  const int wave = t >> 5;
  const int lane = t & 31;
  const int mt   = wave & 3;            // 0..3  : 16-row tile
  const int ntB  = (wave >> 2) * 2;     // 0 or 2: pair of 16-col tiles

  v8f zero8 = {0.f, 0.f, 0.f, 0.f, 0.f, 0.f, 0.f, 0.f};
  v8f acc[2];
  acc[0] = zero8;
  acc[1] = zero8;

  const int am = t >> 2;          // 0..63 (A row within tile)
  const int ak = (t & 3) * 8;     // 0,8,16,24
  const int bk = t >> 3;          // 0..31 (B row within K tile)
  const int bn = (t & 7) * 8;     // 0..56

  for (int k0 = 0; k0 < K; k0 += 32) {
    // ---- stage A tile (64x32) into LDS as bf16 ----
    if (A_BF16) {
      const __bf16* A = (const __bf16*)Aptr;
      // async global->LDS b128 copy (falls back to load+store)
      ASYNC_B128(A + (size_t)(m0 + am) * K + k0 + ak, &aT[am * 40 + ak]);
    } else {
      const float* A = (const float*)Aptr;
      const float4 v0 = *(const float4*)(A + (size_t)(m0 + am) * K + k0 + ak);
      const float4 v1 = *(const float4*)(A + (size_t)(m0 + am) * K + k0 + ak + 4);
      __bf16* dst = &aT[am * 40 + ak];
      dst[0] = (__bf16)v0.x; dst[1] = (__bf16)v0.y; dst[2] = (__bf16)v0.z; dst[3] = (__bf16)v0.w;
      dst[4] = (__bf16)v1.x; dst[5] = (__bf16)v1.y; dst[6] = (__bf16)v1.z; dst[7] = (__bf16)v1.w;
    }
    // ---- stage B tile (32x64) transposed into LDS ----
    {
      uint4 v = *(const uint4*)(Bw + (size_t)(k0 + bk) * Nout + n0 + bn);
      const __bf16* hv = (const __bf16*)&v;
#pragma unroll
      for (int i = 0; i < 8; ++i) bT[(bn + i) * 40 + bk] = hv[i];
    }
    // prefetch next A tile chunk (global_prefetch_b8)
    if (k0 + 32 < K) {
      if (A_BF16)
        __builtin_prefetch((const __bf16*)Aptr + (size_t)(m0 + am) * K + k0 + 32 + ak, 0, 1);
      else
        __builtin_prefetch((const float*)Aptr + (size_t)(m0 + am) * K + k0 + 32 + ak, 0, 1);
    }
    if (A_BF16) ASYNC_WAIT();
    __syncthreads();

    v16bf a  = load_frag_a(&aT[mt * 16 * 40], 40, lane);
    v16bf b0 = load_frag_b(&bT[(ntB + 0) * 16 * 40], 40, lane);
    v16bf b1 = load_frag_b(&bT[(ntB + 1) * 16 * 40], 40, lane);
    acc[0] = __builtin_amdgcn_wmma_f32_16x16x32_bf16(false, a, false, b0, (short)0, acc[0], false, false);
    acc[1] = __builtin_amdgcn_wmma_f32_16x16x32_bf16(false, a, false, b1, (short)0, acc[1], false, false);

    __syncthreads();
  }

  // ---- epilogue: D layout (M = r + 8*(lane>=16), N = lane&15) ----
#pragma unroll
  for (int j = 0; j < 2; ++j) {
    const int n = n0 + (ntB + j) * 16 + (lane & 15);
    const float bv = bias[n];
#pragma unroll
    for (int r = 0; r < 8; ++r) {
      const int m = m0 + mt * 16 + r + ((lane >> 4) << 3);
      const float val = acc[j][r] + bv;
      if (OUT_BF16)
        ((__bf16*)Outp)[(size_t)m * Nout + n] = (__bf16)val;
      else
        ((float*)Outp)[(size_t)m * Nout + n] = val;
    }
  }
}

// ---------------------------------------------------------------------------
// Fused window attention: one block per (window b, head h).
// qkv: bf16 [B*49, 768]  ->  out: bf16 [B*49, 256]
// cb: combined bias [64][8][49*49] fp32
// ---------------------------------------------------------------------------
__global__ __launch_bounds__(128)
void attn_kernel(const __bf16* __restrict__ qkv, const float* __restrict__ cb,
                 __bf16* __restrict__ outbuf) {
  __shared__ alignas(16) __bf16 qs [64 * 40];  // q rows  [64][32], stride 40
  __shared__ alignas(16) __bf16 ks [64 * 40];  // k rows  [64][32] (== B^T for QK^T)
  __shared__ alignas(16) __bf16 vts[32 * 80];  // v^T     [32][64], stride 80
  __shared__ alignas(16) float  sm [64 * 68];  // scores  [64][64], stride 68
  __shared__ alignas(16) __bf16 ps [64 * 80];  // softmax [64][64], stride 80

  const int t    = threadIdx.x;
  const int b    = blockIdx.x >> 3;
  const int h    = blockIdx.x & 7;
  const int w    = b & (NWIN - 1);         // window index = b % NUM_WINDOWS
  const int wave = t >> 5;
  const int lane = t & 31;

  // ---- load q/k/v tiles; pad rows 49..63 with zeros ----
  const int row = t >> 1;      // 0..63
  const int hh  = t & 1;       // which 16-half of the 32-wide head dim
  uint4 zero; zero.x = zero.y = zero.z = zero.w = 0u;
  uint4 vv = zero;
  if (row < N_TOK) {
    const __bf16* src = qkv + ((size_t)b * N_TOK + row) * QKV_COLS + h * HDIM + hh * 16;
    ASYNC_B128(src,       &qs[row * 40 + hh * 16]);
    ASYNC_B128(src + 256, &ks[row * 40 + hh * 16]);
    vv = *(const uint4*)(src + 512);
  } else {
    *(uint4*)(&qs[row * 40 + hh * 16]) = zero;
    *(uint4*)(&ks[row * 40 + hh * 16]) = zero;
  }
  {
    const __bf16* hv = (const __bf16*)&vv;
#pragma unroll
    for (int i = 0; i < 16; ++i) vts[(hh * 16 + i) * 80 + row] = hv[i];
  }
  ASYNC_WAIT();
  __syncthreads();

  // ---- S = q @ k^T : branch-free epilogue, raw scaled scores to LDS ----
  {
    v16bf aq = load_frag_a(&qs[wave * 16 * 40], 40, lane);
#pragma unroll
    for (int nt = 0; nt < 4; ++nt) {
      v16bf bk = load_frag_b(&ks[nt * 16 * 40], 40, lane);
      v8f acc = {0.f, 0.f, 0.f, 0.f, 0.f, 0.f, 0.f, 0.f};
      acc = __builtin_amdgcn_wmma_f32_16x16x32_bf16(false, aq, false, bk, (short)0, acc, false, false);
      const int jj = nt * 16 + (lane & 15);
#pragma unroll
      for (int r = 0; r < 8; ++r) {
        const int ii = wave * 16 + r + ((lane >> 4) << 3);
        sm[ii * 68 + jj] = acc[r] * ATT_SCALE;
      }
    }
  }
  __syncthreads();

  // ---- softmax over valid 49 columns (bias added here); P zero-padded ----
  if (t < 64) {
    float*  srow = &sm[t * 68];
    __bf16* prow = &ps[t * 80];
    if (t < N_TOK) {
      const float* cbrow = cb + ((size_t)(w * NHEAD + h)) * NN + t * N_TOK;
      float mx = -3.4e38f;
      for (int j = 0; j < N_TOK; ++j) {
        const float sv = srow[j] + cbrow[j];
        srow[j] = sv;
        mx = fmaxf(mx, sv);
      }
      float s = 0.f;
      for (int j = 0; j < N_TOK; ++j) { float e = __expf(srow[j] - mx); srow[j] = e; s += e; }
      const float inv = 1.0f / s;
      for (int j = 0; j < N_TOK; ++j) prow[j] = (__bf16)(srow[j] * inv);
      for (int j = N_TOK; j < 64; ++j) prow[j] = (__bf16)0.f;
    } else {
      for (int j = 0; j < 64; ++j) prow[j] = (__bf16)0.f;
    }
  }
  __syncthreads();

  // ---- O = P @ V  (K = 64 -> two K-steps of 32; N = 32 -> 2 col tiles) ----
  v8f zero8 = {0.f, 0.f, 0.f, 0.f, 0.f, 0.f, 0.f, 0.f};
  v8f oacc[2];
  oacc[0] = zero8;
  oacc[1] = zero8;
#pragma unroll
  for (int k0 = 0; k0 < 64; k0 += 32) {
    v16bf ap = load_frag_a(&ps[wave * 16 * 80 + k0], 80, lane);
#pragma unroll
    for (int nt = 0; nt < 2; ++nt) {
      v16bf bv = load_frag_b(&vts[nt * 16 * 80 + k0], 80, lane);
      oacc[nt] = __builtin_amdgcn_wmma_f32_16x16x32_bf16(false, ap, false, bv, (short)0, oacc[nt], false, false);
    }
  }
#pragma unroll
  for (int nt = 0; nt < 2; ++nt) {
#pragma unroll
    for (int r = 0; r < 8; ++r) {
      const int ii = wave * 16 + r + ((lane >> 4) << 3);
      const int dd = nt * 16 + (lane & 15);
      if (ii < N_TOK)
        outbuf[((size_t)b * N_TOK + ii) * DIM_ + h * HDIM + dd] = (__bf16)oacc[nt][r];
    }
  }
}

// ---------------------------------------------------------------------------
// launch
// ---------------------------------------------------------------------------
extern "C" void kernel_launch(void* const* d_in, const int* in_sizes, int n_in,
                              void* d_out, int out_size, void* d_ws, size_t ws_size,
                              hipStream_t stream) {
  (void)in_sizes; (void)n_in; (void)out_size; (void)ws_size;
  const float* x          = (const float*)d_in[0];   // [2048, 49, 256]
  const float* mask       = (const float*)d_in[1];   // [64, 49, 49]
  const float* qkv_w      = (const float*)d_in[2];   // [256, 768]
  const float* qkv_b      = (const float*)d_in[3];   // [768]
  const float* proj_w     = (const float*)d_in[4];   // [256, 256]
  const float* proj_b     = (const float*)d_in[5];   // [256]
  const float* bias_table = (const float*)d_in[6];   // [169, 8]
  const int*   rel_index  = (const int*)d_in[7];     // [49, 49]

  char* ws = (char*)d_ws;
  __bf16* qkvw_bf  = (__bf16*)(ws);                                // 393,216 B
  __bf16* projw_bf = (__bf16*)(ws + 393216);                       // 131,072 B
  __bf16* qkvbuf   = (__bf16*)(ws + 524288);                       // 100352*768*2
  __bf16* attnbuf  = (__bf16*)(ws + 524288 + (size_t)100352 * 768 * 2);
  float*  cbbuf    = (float*)(ws + 524288 + (size_t)100352 * 768 * 2
                                         + (size_t)100352 * 256 * 2);

  cvt_f32_bf16<<<(256 * 768 + 255) / 256, 256, 0, stream>>>(qkv_w, qkvw_bf, 256 * 768);
  cvt_f32_bf16<<<(256 * 256 + 255) / 256, 256, 0, stream>>>(proj_w, projw_bf, 256 * 256);
  build_bias<<<(NWIN * NHEAD * NN + 255) / 256, 256, 0, stream>>>(bias_table, rel_index,
                                                                  mask, cbbuf);

  // QKV projection: [100352,256] x [256,768] -> bf16
  dim3 g1(QKV_COLS / 64, (BTOT * N_TOK) / 64);   // (12, 1568)
  gemm64<false, true><<<g1, 256, 0, stream>>>(x, qkvw_bf, qkv_b, qkvbuf, DIM_, QKV_COLS);

  // fused attention per (window, head)
  attn_kernel<<<BTOT * NHEAD, 128, 0, stream>>>(qkvbuf, cbbuf, attnbuf);

  // output projection: [100352,256] x [256,256] -> fp32 d_out
  dim3 g3(DIM_ / 64, (BTOT * N_TOK) / 64);       // (4, 1568)
  gemm64<true, false><<<g3, 256, 0, stream>>>(attnbuf, projw_bf, proj_b, d_out, DIM_, DIM_);
}
